// Seq2Seq_61735859913191
// MI455X (gfx1250) — matrix operand order repaired and checked by reading, not compiled, not run on hardware
//
#include <hip/hip_runtime.h>
#include <hip/hip_bf16.h>

#define SRC_LEN 64
#define TRG_LEN 64
#define BATCH   64
#define EDIM    256
#define HDIM    512
#define G4H     2048   // 4*H
#define OUTV    6000
#define OUTV_PAD 6016  // padded to multiple of 64 (4 N-tiles per wave)

typedef __attribute__((ext_vector_type(16))) __bf16 v16bf;
typedef __attribute__((ext_vector_type(8)))  __bf16 v8bf;
typedef __attribute__((ext_vector_type(8)))  float  v8f;
typedef __hip_bfloat16 bf16_t;

// ---------------- utility kernels ----------------

__global__ void k_zero_f32(float* p, int n) {
  int i = blockIdx.x * blockDim.x + threadIdx.x;
  if (i < n) p[i] = 0.0f;
}

__global__ void k_zero_bf16(bf16_t* p, int n) {
  int i = blockIdx.x * blockDim.x + threadIdx.x;
  if (i < n) p[i] = __float2bfloat16(0.0f);
}

// pack f32 [rows][sc] -> bf16 dst[r][doff + c] with row stride dstride
__global__ void k_pack_bf16(const float* __restrict__ src, bf16_t* __restrict__ dst,
                            int rows, int sc, int dstride, int doff) {
  int i = blockIdx.x * blockDim.x + threadIdx.x;
  if (i >= rows * sc) return;
  int r = i / sc, c = i - r * sc;
  dst[r * dstride + doff + c] = __float2bfloat16(src[i]);
}

// gather embedding rows for B tokens into bf16 activation slot
__global__ void k_embed(const float* __restrict__ emb, const int* __restrict__ tok,
                        bf16_t* __restrict__ dst, int dstride, int doff) {
  int i = blockIdx.x * blockDim.x + threadIdx.x;
  if (i >= BATCH * EDIM) return;
  int b = i / EDIM, e = i - b * EDIM;
  dst[b * dstride + doff + e] = __float2bfloat16(emb[(long)tok[b] * EDIM + e]);
}

// ---------------- WMMA NT GEMM: Out[m,n] = sum_k A[m,k]*W[n,k] (+bias[n]) ----------------
// A bf16 [64][lda], W bf16 [>=Ntiles*16][K] row-major, Out f32 [64][N].
// Each wave owns one M-tile (of 4) x 4 consecutive N-tiles; A fragment is loaded once
// per k-step and reused for 4 WMMAs (2 A-loads + 8 B-loads -> 4 WMMAs).
__global__ void k_gemm_bf16_nt(const bf16_t* __restrict__ A, int lda,
                               const bf16_t* __restrict__ W, int K, int N,
                               float* __restrict__ Out,
                               const float* __restrict__ bias,
                               int Ngroups, int Ntiles) {
  const int wave = threadIdx.x >> 5;
  const int lane = threadIdx.x & 31;
  const int task = blockIdx.x * (blockDim.x >> 5) + wave;
  if (task >= 4 * Ngroups) return;          // wave-uniform exit (EXEC stays all-1)
  const int mt = task & 3;
  const int ng = task >> 2;                 // group of 4 N-tiles
  const int r16 = lane & 15;
  const int hi  = lane >> 4;

  // A fragment: lanes 0-15 hold K k..k+15; lanes 16-31 K k+16..k+31
  const __bf16* a = reinterpret_cast<const __bf16*>(A) + (long)(mt * 16 + r16) * lda + hi * 16;
  // B fragments: lane = row of W; lanes 0-15 K=k..k+15, lanes 16-31 K=k+16..k+31
  const __bf16* b = reinterpret_cast<const __bf16*>(W) + (long)(ng * 64 + r16) * K + hi * 16;
  const long bstep = (long)16 * K;          // stride between adjacent N-tiles

  v8f acc0 = {}, acc1 = {}, acc2 = {}, acc3 = {};
  for (int k = 0; k < K; k += 32) {
    v8bf a0 = *reinterpret_cast<const v8bf*>(a + k);
    v8bf a1 = *reinterpret_cast<const v8bf*>(a + k + 8);
    v16bf av = __builtin_shufflevector(a0, a1, 0, 1, 2, 3, 4, 5, 6, 7,
                                               8, 9, 10, 11, 12, 13, 14, 15);
    v16bf bv0 = *reinterpret_cast<const v16bf*>(b + k);
    v16bf bv1 = *reinterpret_cast<const v16bf*>(b + bstep + k);
    v16bf bv2 = *reinterpret_cast<const v16bf*>(b + 2 * bstep + k);
    v16bf bv3 = *reinterpret_cast<const v16bf*>(b + 3 * bstep + k);
    acc0 = __builtin_amdgcn_wmma_f32_16x16x32_bf16(false, av, false, bv0, (short)0, acc0, false, false);
    acc1 = __builtin_amdgcn_wmma_f32_16x16x32_bf16(false, av, false, bv1, (short)0, acc1, false, false);
    acc2 = __builtin_amdgcn_wmma_f32_16x16x32_bf16(false, av, false, bv2, (short)0, acc2, false, false);
    acc3 = __builtin_amdgcn_wmma_f32_16x16x32_bf16(false, av, false, bv3, (short)0, acc3, false, false);
  }

  v8f accs[4] = {acc0, acc1, acc2, acc3};
#pragma unroll
  for (int j = 0; j < 4; ++j) {
    int nt = ng * 4 + j;
    if (nt >= Ntiles) break;                // wave-uniform guard for padded tail
    int col = nt * 16 + r16;
    float ba = bias ? bias[col] : 0.0f;
#pragma unroll
    for (int r = 0; r < 8; ++r) {
      int orow = mt * 16 + r + 8 * hi;      // C/D layout: M = vgpr + 8*(lane>=16)
      Out[(long)orow * N + col] = accs[j][r] + ba;
    }
  }
}

// ---------------- LSTM cell (pointwise) ----------------
__device__ __forceinline__ float sigm(float x) { return 1.0f / (1.0f + __expf(-x)); }

// gate order i,f,g,o; writes updated c,h (f32) + bf16 copies of h into up to 2 slots
__global__ void k_lstm_cell(const float* __restrict__ G,
                            const float* __restrict__ bih, const float* __restrict__ bhh,
                            float* __restrict__ c, float* __restrict__ h,
                            bf16_t* dA, int sA, int oA,
                            bf16_t* dB, int sB, int oB,
                            float* extra) {
  int i = blockIdx.x * blockDim.x + threadIdx.x;
  if (i >= BATCH * HDIM) return;
  int b = i >> 9, hh = i & (HDIM - 1);
  const float* g = G + b * G4H;
  float gi = g[hh]             + bih[hh]             + bhh[hh];
  float gf = g[HDIM + hh]      + bih[HDIM + hh]      + bhh[HDIM + hh];
  float gg = g[2 * HDIM + hh]  + bih[2 * HDIM + hh]  + bhh[2 * HDIM + hh];
  float go = g[3 * HDIM + hh]  + bih[3 * HDIM + hh]  + bhh[3 * HDIM + hh];
  float cn = sigm(gf) * c[i] + sigm(gi) * tanhf(gg);
  float hn = sigm(go) * tanhf(cn);
  c[i] = cn;
  h[i] = hn;
  bf16_t hb = __float2bfloat16(hn);
  if (dA) dA[b * sA + oA + hh] = hb;
  if (dB) dB[b * sB + oB + hh] = hb;
  if (extra) extra[i] = hn;
}

// ---------------- attention ----------------
// scores[b][s] = dot(enc_outs[s][b][:], h1[b][:]) ; softmax over s -> attn[b][s]
__global__ void k_attn(const float* __restrict__ enc_outs, const float* __restrict__ h1,
                       float* __restrict__ attn) {
  __shared__ float sc[SRC_LEN];
  __shared__ float s_m, s_s;
  int b = blockIdx.x, s = threadIdx.x;
  const float* eo = enc_outs + ((long)s * BATCH + b) * HDIM;
  const float* hv = h1 + (long)b * HDIM;
  float d = 0.0f;
  for (int k = 0; k < HDIM; ++k) d += eo[k] * hv[k];
  sc[s] = d;
  __syncthreads();
  if (s == 0) {
    float m = sc[0];
    for (int j = 1; j < SRC_LEN; ++j) m = fmaxf(m, sc[j]);
    float t = 0.0f;
    for (int j = 0; j < SRC_LEN; ++j) t += __expf(sc[j] - m);
    s_m = m; s_s = t;
  }
  __syncthreads();
  attn[b * SRC_LEN + s] = __expf(d - s_m) / s_s;
}

// weighted[b][h] = sum_s attn[b][s] * enc_outs[s][b][h] -> bf16 into xd0 slot
__global__ void k_weighted(const float* __restrict__ enc_outs, const float* __restrict__ attn,
                           bf16_t* __restrict__ dst, int dstride, int doff) {
  int i = blockIdx.x * blockDim.x + threadIdx.x;
  if (i >= BATCH * HDIM) return;
  int b = i >> 9, hh = i & (HDIM - 1);
  float w = 0.0f;
  for (int s = 0; s < SRC_LEN; ++s)
    w += attn[b * SRC_LEN + s] * enc_outs[((long)s * BATCH + b) * HDIM + hh];
  dst[b * dstride + doff + hh] = __float2bfloat16(w);
}

// ---------------- host orchestration ----------------

extern "C" void kernel_launch(void* const* d_in, const int* in_sizes, int n_in,
                              void* d_out, int out_size, void* d_ws, size_t ws_size,
                              hipStream_t stream) {
  (void)in_sizes; (void)n_in; (void)out_size; (void)ws_size;

  const int*   src      = (const int*)  d_in[0];
  const int*   trg      = (const int*)  d_in[1];
  const float* enc_emb  = (const float*)d_in[2];
  const float* enc_Wih0 = (const float*)d_in[3];
  const float* enc_Whh0 = (const float*)d_in[4];
  const float* enc_bih0 = (const float*)d_in[5];
  const float* enc_bhh0 = (const float*)d_in[6];
  const float* enc_Wih1 = (const float*)d_in[7];
  const float* enc_Whh1 = (const float*)d_in[8];
  const float* enc_bih1 = (const float*)d_in[9];
  const float* enc_bhh1 = (const float*)d_in[10];
  const float* dec_emb  = (const float*)d_in[11];
  const float* dec_Wih0 = (const float*)d_in[12];
  const float* dec_Whh0 = (const float*)d_in[13];
  const float* dec_bih0 = (const float*)d_in[14];
  const float* dec_bhh0 = (const float*)d_in[15];
  const float* dec_Wih1 = (const float*)d_in[16];
  const float* dec_Whh1 = (const float*)d_in[17];
  const float* dec_bih1 = (const float*)d_in[18];
  const float* dec_bhh1 = (const float*)d_in[19];
  const float* fc_W     = (const float*)d_in[20];
  const float* fc_b     = (const float*)d_in[21];
  float* out = (float*)d_out;

  // ---- carve workspace (256B aligned slabs) ----
  char* p = (char*)d_ws;
  auto carve = [&](size_t bytes) -> char* {
    char* r = p;
    p += (bytes + 255) & ~(size_t)255;
    return r;
  };
  bf16_t* We0 = (bf16_t*)carve((size_t)G4H * 768  * 2);     // [enc_Wih0 | enc_Whh0]
  bf16_t* We1 = (bf16_t*)carve((size_t)G4H * 1024 * 2);     // [enc_Wih1 | enc_Whh1]
  bf16_t* Wd0 = (bf16_t*)carve((size_t)G4H * 1280 * 2);     // [dec_Wih0 | dec_Whh0]
  bf16_t* Wd1 = (bf16_t*)carve((size_t)G4H * 1024 * 2);     // [dec_Wih1 | dec_Whh1]
  bf16_t* Wfc = (bf16_t*)carve((size_t)OUTV_PAD * HDIM * 2);
  bf16_t* xe0 = (bf16_t*)carve((size_t)BATCH * 768  * 2);   // [emb | h0]
  bf16_t* xe1 = (bf16_t*)carve((size_t)BATCH * 1024 * 2);   // [h0 | h1]
  bf16_t* xd0 = (bf16_t*)carve((size_t)BATCH * 1280 * 2);   // [emb | weighted | h0]
  bf16_t* xd1 = (bf16_t*)carve((size_t)BATCH * 1024 * 2);   // [h0 | h1]
  float* h0 = (float*)carve((size_t)BATCH * HDIM * 4);
  float* c0 = (float*)carve((size_t)BATCH * HDIM * 4);
  float* h1 = (float*)carve((size_t)BATCH * HDIM * 4);
  float* c1 = (float*)carve((size_t)BATCH * HDIM * 4);
  float* G  = (float*)carve((size_t)BATCH * G4H * 4);
  float* enc_outs = (float*)carve((size_t)SRC_LEN * BATCH * HDIM * 4); // [s][b][h]
  float* attn = (float*)carve((size_t)BATCH * SRC_LEN * 4);

  auto blocks = [](int n) { return (n + 255) / 256; };
  auto pack = [&](const float* s, bf16_t* d, int rows, int sc, int ds, int off) {
    k_pack_bf16<<<blocks(rows * sc), 256, 0, stream>>>(s, d, rows, sc, ds, off);
  };
  auto gemm = [&](const bf16_t* A, int lda, const bf16_t* W, int K, int N,
                  float* Out, const float* bias) {
    int ntiles = N / 16;
    int ngroups = (ntiles + 3) / 4;
    int tasks = 4 * ngroups;                 // 4 M-tiles x N-tile-groups
    int nblk = (tasks + 7) / 8;              // 8 waves per 256-thread block
    k_gemm_bf16_nt<<<nblk, 256, 0, stream>>>(A, lda, W, K, N, Out, bias, ngroups, ntiles);
  };
  auto cell = [&](const float* bih, const float* bhh, float* cc, float* hh,
                  bf16_t* dA, int sA, int oA, bf16_t* dB, int sB, int oB, float* ex) {
    k_lstm_cell<<<blocks(BATCH * HDIM), 256, 0, stream>>>(G, bih, bhh, cc, hh,
                                                          dA, sA, oA, dB, sB, oB, ex);
  };

  // ---- pack weights to bf16 (K-concatenated NT layout) ----
  pack(enc_Wih0, We0, G4H, EDIM, 768, 0);
  pack(enc_Whh0, We0, G4H, HDIM, 768, EDIM);
  pack(enc_Wih1, We1, G4H, HDIM, 1024, 0);
  pack(enc_Whh1, We1, G4H, HDIM, 1024, HDIM);
  pack(dec_Wih0, Wd0, G4H, EDIM + HDIM, 1280, 0);
  pack(dec_Whh0, Wd0, G4H, HDIM, 1280, EDIM + HDIM);
  pack(dec_Wih1, Wd1, G4H, HDIM, 1024, 0);
  pack(dec_Whh1, Wd1, G4H, HDIM, 1024, HDIM);
  pack(fc_W,     Wfc, OUTV, HDIM, HDIM, 0);
  // zero the padded tail rows of Wfc (read by last tile-group, never stored)
  k_zero_bf16<<<blocks((OUTV_PAD - OUTV) * HDIM), 256, 0, stream>>>(
      Wfc + (size_t)OUTV * HDIM, (OUTV_PAD - OUTV) * HDIM);

  // ---- init: zero states, zero activation slabs, zero output step 0 ----
  k_zero_f32<<<blocks(4 * BATCH * HDIM), 256, 0, stream>>>(h0, 4 * BATCH * HDIM);
  int xtot = BATCH * (768 + 1024 + 1280 + 1024); // xe0..xd1 are contiguous slabs
  k_zero_bf16<<<blocks(xtot), 256, 0, stream>>>(xe0, xtot);
  k_zero_f32<<<blocks(BATCH * OUTV), 256, 0, stream>>>(out, BATCH * OUTV);

  // ---- encoder: 64 steps ----
  for (int t = 0; t < SRC_LEN; ++t) {
    k_embed<<<blocks(BATCH * EDIM), 256, 0, stream>>>(enc_emb, src + t * BATCH, xe0, 768, 0);
    gemm(xe0, 768, We0, 768, G4H, G, nullptr);
    cell(enc_bih0, enc_bhh0, c0, h0, xe0, 768, EDIM, xe1, 1024, 0, nullptr);
    gemm(xe1, 1024, We1, 1024, G4H, G, nullptr);
    cell(enc_bih1, enc_bhh1, c1, h1, xe1, 1024, HDIM, nullptr, 0, 0,
         enc_outs + (size_t)t * BATCH * HDIM);
  }

  // ---- seed decoder activation slots with encoder-final h0/h1 ----
  pack(h0, xd0, BATCH, HDIM, 1280, EDIM + HDIM);
  pack(h1, xd1, BATCH, HDIM, 1024, HDIM);

  // ---- decoder: 63 steps (teacher forcing over trg[:-1]) ----
  for (int t = 0; t < TRG_LEN - 1; ++t) {
    k_attn<<<BATCH, SRC_LEN, 0, stream>>>(enc_outs, h1, attn);
    k_weighted<<<blocks(BATCH * HDIM), 256, 0, stream>>>(enc_outs, attn, xd0, 1280, EDIM);
    k_embed<<<blocks(BATCH * EDIM), 256, 0, stream>>>(dec_emb, trg + t * BATCH, xd0, 1280, 0);
    gemm(xd0, 1280, Wd0, 1280, G4H, G, nullptr);
    cell(dec_bih0, dec_bhh0, c0, h0, xd1, 1024, 0, xd0, 1280, EDIM + HDIM, nullptr);
    gemm(xd1, 1024, Wd1, 1024, G4H, G, nullptr);
    cell(dec_bih1, dec_bhh1, c1, h1, xd1, 1024, HDIM, nullptr, 0, 0, nullptr);
    // pred = h1 @ fc_W.T + fc_b  -> out[t+1]
    gemm(xd1 + HDIM, 1024, Wfc, HDIM, OUTV,
         out + (size_t)(t + 1) * BATCH * OUTV, fc_b);
  }
}
